// myRNN_8942121910459
// MI455X (gfx1250) — matrix operand order repaired and checked by reading, not compiled, not run on hardware
//
#include <hip/hip_runtime.h>
#include <math.h>

// ---------------------------------------------------------------------------
// Problem constants (from reference): B=32, T=2048, D=512, H=512, L=2, O=2
// ---------------------------------------------------------------------------
namespace {
constexpr int Bb = 32, Tt = 2048, Dd = 512, Hh = 512, Ll = 2, Oo = 2;
constexpr int GATES = 4 * Hh;             // 2048 gate columns per layer
constexpr int NWG = 16;                   // persistent workgroups (1 per WGP)
constexpr int WPW = 4;                    // waves per workgroup (128 threads)
constexpr int NJ  = Hh / 16;              // 32 h-column tiles

// workspace layout (bytes)
constexpr size_t SZ_W    = (size_t)Ll * GATES * Dd * sizeof(_Float16); // 4 MiB
constexpr size_t OFF_WXH = 0;
constexpr size_t OFF_WHH = OFF_WXH + SZ_W;
constexpr size_t OFF_H   = OFF_WHH + SZ_W;            // h[layer][parity][B][H] f16
constexpr size_t SZ_H    = (size_t)Ll * 2 * Bb * Hh * sizeof(_Float16);
constexpr size_t OFF_PRT = OFF_H + SZ_H;              // part[T][NJ][B][O] f32
constexpr size_t SZ_PRT  = (size_t)Tt * NJ * Bb * Oo * sizeof(float);
constexpr size_t OFF_BAR = OFF_PRT + SZ_PRT;          // 2 x u32 barrier state
constexpr size_t OFF_X16 = ((OFF_BAR + 256) + 255) & ~(size_t)255; // x in f16
constexpr size_t SZ_X16  = (size_t)Bb * Tt * Dd * sizeof(_Float16); // 64 MiB
}

typedef __attribute__((ext_vector_type(16))) _Float16 v16h;
typedef __attribute__((ext_vector_type(8)))  float    v8f;
typedef __attribute__((ext_vector_type(4)))  unsigned u32x4;

// ---------------------------------------------------------------------------
// WMMA wrapper: D = A(16x32 f16) * B(32x16 f16) + C(16x16 f32)
// ---------------------------------------------------------------------------
__device__ __forceinline__ v8f wmma_f16(v16h a, v16h b, v8f c) {
    return __builtin_amdgcn_wmma_f32_16x16x32_f16(
        /*neg_a=*/false, a, /*neg_b=*/false, b,
        /*c_mod=*/(short)0, c, /*reuse_a=*/false, /*reuse_b=*/false);
}

// A-fragment (16x32 f16, ISA 7.12.2 layout) from row-major f16, row stride ld.
// lane l<16 : row=l,    halves 0..7 = K 0..7,   halves 8..15 = K 16..23
// lane l>=16: row=l-16, halves 0..7 = K 8..15,  halves 8..15 = K 24..31
__device__ __forceinline__ v16h ld_fragA16(const _Float16* base, size_t ld,
                                           int row0, int k0) {
    const int l = threadIdx.x & 31;
    const _Float16* q = base + (size_t)(row0 + (l & 15)) * ld + k0 + ((l >> 4) << 3);
    union { u32x4 u[2]; v16h h; } c;
    c.u[0] = *reinterpret_cast<const u32x4*>(q);        // K +0..7
    c.u[1] = *reinterpret_cast<const u32x4*>(q + 16);   // K +16..23
    return c.h;
}

// B-fragment (32x16 f16): lane l holds column n = n0 + (l&15);
// lanes 0-15 carry K 0..7,16..23, lanes 16-31 carry K 8..15,24..31
// (same k order as A; weights stored [n][k] row-major).
__device__ __forceinline__ v16h ld_fragB16(const _Float16* base, size_t ld,
                                           int n0, int k0) {
    const int l = threadIdx.x & 31;
    const _Float16* q = base + (size_t)(n0 + (l & 15)) * ld + k0 + ((l >> 4) << 3);
    union { u32x4 u[2]; v16h h; } c;
    c.u[0] = *reinterpret_cast<const u32x4*>(q);        // K +0..7
    c.u[1] = *reinterpret_cast<const u32x4*>(q + 16);   // K +16..23
    return c.h;
}

__device__ __forceinline__ float sigmoidf_(float v) {
    return 1.0f / (1.0f + __expf(-v));
}

// sense-reversal grid barrier across NWG workgroups (state in d_ws)
__device__ __forceinline__ void grid_barrier(unsigned* cnt, unsigned* gen,
                                             unsigned nwg) {
    __syncthreads();
    if (threadIdx.x == 0) {
        __threadfence();
        unsigned g0 = __hip_atomic_load(gen, __ATOMIC_RELAXED, __HIP_MEMORY_SCOPE_AGENT);
        unsigned a  = __hip_atomic_fetch_add(cnt, 1u, __ATOMIC_ACQ_REL, __HIP_MEMORY_SCOPE_AGENT);
        if (a == nwg - 1u) {
            __hip_atomic_store(cnt, 0u, __ATOMIC_RELAXED, __HIP_MEMORY_SCOPE_AGENT);
            __hip_atomic_fetch_add(gen, 1u, __ATOMIC_RELEASE, __HIP_MEMORY_SCOPE_AGENT);
        } else {
            while (__hip_atomic_load(gen, __ATOMIC_ACQUIRE, __HIP_MEMORY_SCOPE_AGENT) == g0) {
                __builtin_amdgcn_s_sleep(8);
            }
        }
        __threadfence();
    }
    __syncthreads();
}

// ---------------------------------------------------------------------------
// Prolog: fp32 -> f16 conversion of weights AND x, + barrier state init.
// Fully parallel, HBM-bound (~200 MB moved once).
// ---------------------------------------------------------------------------
__global__ void lstm_prep_kernel(const float* __restrict__ x,
                                 const float* __restrict__ Wxh,
                                 const float* __restrict__ Whh,
                                 char* __restrict__ ws) {
    _Float16* wx  = reinterpret_cast<_Float16*>(ws + OFF_WXH);
    _Float16* wh  = reinterpret_cast<_Float16*>(ws + OFF_WHH);
    _Float16* x16 = reinterpret_cast<_Float16*>(ws + OFF_X16);
    const size_t nw = (size_t)Ll * GATES * Dd;
    const size_t nx = (size_t)Bb * Tt * Dd;
    const size_t stride = (size_t)gridDim.x * blockDim.x;
    for (size_t i = (size_t)blockIdx.x * blockDim.x + threadIdx.x; i < nx; i += stride) {
        x16[i] = (_Float16)x[i];
        if (i < nw) {
            wx[i] = (_Float16)Wxh[i];
            wh[i] = (_Float16)Whh[i];
        }
    }
    if (blockIdx.x == 0 && threadIdx.x < 2)
        reinterpret_cast<unsigned*>(ws + OFF_BAR)[threadIdx.x] = 0u;
}

// ---------------------------------------------------------------------------
// Persistent LSTM: 16 WGs x 4 waves = 64 waves. Wave w owns the 16x16 tile
// (m-tile = w&1, h-col tile j = w>>1) of h/c for both layers; c stays in VGPRs.
// Per step: 4 gate tiles x (K=512 x-gemm + K=512 h-gemm) per layer via WMMA.
// h state is exchanged through L2 with 2 grid barriers per step.
// ---------------------------------------------------------------------------
__global__ void __launch_bounds__(WPW * 32)
lstm_persistent_kernel(const float* __restrict__ bxh,
                       const float* __restrict__ bhh,
                       const float* __restrict__ wreg,
                       char* __restrict__ ws) {
    const _Float16* Wxh16 = reinterpret_cast<const _Float16*>(ws + OFF_WXH);
    const _Float16* Whh16 = reinterpret_cast<const _Float16*>(ws + OFF_WHH);
    const _Float16* x16   = reinterpret_cast<const _Float16*>(ws + OFF_X16);
    _Float16* Hbuf = reinterpret_cast<_Float16*>(ws + OFF_H);   // [L][parity][B][H]
    float*    part = reinterpret_cast<float*>(ws + OFF_PRT);    // [T][NJ][B][O]
    unsigned* bar  = reinterpret_cast<unsigned*>(ws + OFF_BAR);

    const int lane = threadIdx.x & 31;
    const int wid  = threadIdx.x >> 5;
    const int w    = blockIdx.x * WPW + wid;   // 0..63
    const int mt   = w & 1;                    // batch tile (b 0..15 / 16..31)
    const int j    = w >> 1;                   // h column tile 0..31
    const int m0   = mt * 16;
    const int l15  = lane & 15;
    const int lg   = lane >> 4;                // half-wave group
    const int hcol = j * 16 + l15;             // this lane's h column

    // hoisted per-lane constants
    float bias0[4], bias1[4];
#pragma unroll
    for (int g = 0; g < 4; ++g) {
        const int n = g * Hh + hcol;
        bias0[g] = bxh[n] + bhh[n];
        bias1[g] = bxh[GATES + n] + bhh[GATES + n];
    }
    const float wr0 = wreg[hcol];
    const float wr1 = wreg[Hh + hcol];

    // zero parity-0 h buffers (this wave's slice) -> initial hidden state
#pragma unroll
    for (int l = 0; l < Ll; ++l)
#pragma unroll
        for (int r = 0; r < 8; ++r) {
            const int b = m0 + r + 8 * lg;
            Hbuf[(size_t)(l * 2 + 0) * Bb * Hh + (size_t)b * Hh + hcol] = (_Float16)0.f;
        }

    // cell state lives in registers for the whole sequence
    float c0[8], c1[8];
#pragma unroll
    for (int r = 0; r < 8; ++r) { c0[r] = 0.f; c1[r] = 0.f; }

    grid_barrier(bar, bar + 1, NWG);

    for (int t = 0; t < Tt; ++t) {
        const int pr = t & 1, pw = pr ^ 1;
        const _Float16* h0r = Hbuf + (size_t)(0 * 2 + pr) * Bb * Hh;
        _Float16*       h0w = Hbuf + (size_t)(0 * 2 + pw) * Bb * Hh;
        const _Float16* h1r = Hbuf + (size_t)(1 * 2 + pr) * Bb * Hh;
        _Float16*       h1w = Hbuf + (size_t)(1 * 2 + pw) * Bb * Hh;

        // prefetch next timestep's x rows into cache (global_prefetch_b8)
        if (t + 1 < Tt) {
            const _Float16* nx = x16 + (size_t)(m0 + l15) * Tt * Dd + (size_t)(t + 1) * Dd;
            __builtin_prefetch(nx, 0, 1);
        }

        // ================= layer 0 =================
        v8f acc[4];
#pragma unroll
        for (int g = 0; g < 4; ++g)
#pragma unroll
            for (int r = 0; r < 8; ++r) acc[g][r] = bias0[g];

        const _Float16* xt = x16 + (size_t)t * Dd;   // row b at +b*T*D
#pragma unroll 2
        for (int kt = 0; kt < Dd / 32; ++kt) {  // x-gemm: K = D = 512
            v16h a = ld_fragA16(xt, (size_t)Tt * Dd, m0, kt * 32);
#pragma unroll
            for (int g = 0; g < 4; ++g)
                acc[g] = wmma_f16(a, ld_fragB16(Wxh16, Dd, g * Hh + j * 16, kt * 32), acc[g]);
        }
#pragma unroll 2
        for (int kt = 0; kt < Hh / 32; ++kt) {  // h-gemm: K = H = 512
            v16h a = ld_fragA16(h0r, Hh, m0, kt * 32);
#pragma unroll
            for (int g = 0; g < 4; ++g)
                acc[g] = wmma_f16(a, ld_fragB16(Whh16, Hh, g * Hh + j * 16, kt * 32), acc[g]);
        }

        float hy0[8];
#pragma unroll
        for (int r = 0; r < 8; ++r) {
            const float ig = sigmoidf_(acc[0][r]);
            const float fg = sigmoidf_(acc[1][r]);
            const float gg = tanhf(acc[2][r]);
            const float og = sigmoidf_(acc[3][r]);
            const float cold = c0[r];
            c0[r]  = cold * fg + ig * gg;
            hy0[r] = og * tanhf(cold);          // NOTE: tanh of OLD cell (faithful)
        }
#pragma unroll
        for (int r = 0; r < 8; ++r) {
            const int b = m0 + r + 8 * lg;
            h0w[(size_t)b * Hh + hcol] = (_Float16)hy0[r];
        }

        grid_barrier(bar, bar + 1, NWG);        // publish h0(t) to all waves

        // ================= layer 1 =================
        const _Float16* Wxh1 = Wxh16 + (size_t)GATES * Dd;
        const _Float16* Whh1 = Whh16 + (size_t)GATES * Hh;
#pragma unroll
        for (int g = 0; g < 4; ++g)
#pragma unroll
            for (int r = 0; r < 8; ++r) acc[g][r] = bias1[g];

#pragma unroll 2
        for (int kt = 0; kt < Hh / 32; ++kt) {  // input gemm: A = h0(t)
            v16h a = ld_fragA16(h0w, Hh, m0, kt * 32);
#pragma unroll
            for (int g = 0; g < 4; ++g)
                acc[g] = wmma_f16(a, ld_fragB16(Wxh1, Hh, g * Hh + j * 16, kt * 32), acc[g]);
        }
#pragma unroll 2
        for (int kt = 0; kt < Hh / 32; ++kt) {  // recurrent gemm: A = h1(t-1)
            v16h a = ld_fragA16(h1r, Hh, m0, kt * 32);
#pragma unroll
            for (int g = 0; g < 4; ++g)
                acc[g] = wmma_f16(a, ld_fragB16(Whh1, Hh, g * Hh + j * 16, kt * 32), acc[g]);
        }

        float hy1[8];
#pragma unroll
        for (int r = 0; r < 8; ++r) {
            const float ig = sigmoidf_(acc[0][r]);
            const float fg = sigmoidf_(acc[1][r]);
            const float gg = tanhf(acc[2][r]);
            const float og = sigmoidf_(acc[3][r]);
            const float cold = c1[r];
            c1[r]  = cold * fg + ig * gg;
            hy1[r] = og * tanhf(cold);
        }
#pragma unroll
        for (int r = 0; r < 8; ++r) {
            const int b = m0 + r + 8 * lg;
            h1w[(size_t)b * Hh + hcol] = (_Float16)hy1[r];
        }

        // -------- output projection partials: out += hy1 * Wreg^T (O=2) ----
        float p0[8], p1[8];
#pragma unroll
        for (int r = 0; r < 8; ++r) { p0[r] = hy1[r] * wr0; p1[r] = hy1[r] * wr1; }
#pragma unroll
        for (int off = 8; off >= 1; off >>= 1)
#pragma unroll
            for (int r = 0; r < 8; ++r) {
                p0[r] += __shfl_xor(p0[r], off, 16);
                p1[r] += __shfl_xor(p1[r], off, 16);
            }
        if (l15 == 0) {
            const int b0 = m0 + 8 * lg;
            float* pp = part + (((size_t)t * NJ + j) * Bb + b0) * Oo;
#pragma unroll
            for (int r = 0; r < 8; ++r) {
                pp[r * Oo + 0] = p0[r];
                pp[r * Oo + 1] = p1[r];
            }
        }

        grid_barrier(bar, bar + 1, NWG);        // publish h1(t); guard parity swap
    }
}

// ---------------------------------------------------------------------------
// Epilog: deterministic reduction of the 32 per-j partials + breg
// out[b][t][o] = breg[o] + sum_j part[t][j][b][o]
// ---------------------------------------------------------------------------
__global__ void lstm_outreduce_kernel(const char* __restrict__ ws,
                                      const float* __restrict__ breg,
                                      float* __restrict__ out) {
    const float* part = reinterpret_cast<const float*>(ws + OFF_PRT);
    const int i = blockIdx.x * blockDim.x + threadIdx.x;
    if (i >= Bb * Tt * Oo) return;
    const int o = i % Oo;
    const int t = (i / Oo) % Tt;
    const int b = i / (Oo * Tt);
    float s = breg[o];
#pragma unroll 8
    for (int j = 0; j < NJ; ++j)
        s += part[(((size_t)t * NJ + j) * Bb + b) * Oo + o];
    out[((size_t)b * Tt + t) * Oo + o] = s;
}

// ---------------------------------------------------------------------------
// host launcher
// ---------------------------------------------------------------------------
extern "C" void kernel_launch(void* const* d_in, const int* in_sizes, int n_in,
                              void* d_out, int out_size, void* d_ws, size_t ws_size,
                              hipStream_t stream) {
    const float* x    = (const float*)d_in[0];
    const float* Wxh  = (const float*)d_in[1];
    const float* bxh  = (const float*)d_in[2];
    const float* Whh  = (const float*)d_in[3];
    const float* bhh  = (const float*)d_in[4];
    const float* Wreg = (const float*)d_in[5];
    const float* breg = (const float*)d_in[6];
    float* out = (float*)d_out;
    char*  ws  = (char*)d_ws;

    lstm_prep_kernel<<<4096, 256, 0, stream>>>(x, Wxh, Whh, ws);
    lstm_persistent_kernel<<<NWG, WPW * 32, 0, stream>>>(bxh, bhh, Wreg, ws);
    const int nout = Bb * Tt * Oo;
    lstm_outreduce_kernel<<<(nout + 255) / 256, 256, 0, stream>>>(ws, breg, out);
}
